// LayerNormGRUCell_38414187495805
// MI455X (gfx1250) — matrix operand order repaired and checked by reading, not compiled, not run on hardware
//
#include <hip/hip_runtime.h>

typedef __attribute__((ext_vector_type(16))) _Float16 v16h;
typedef __attribute__((ext_vector_type(8)))  _Float16 v8h;
typedef __attribute__((ext_vector_type(8)))  float    v8f;
typedef __attribute__((ext_vector_type(2)))  float    v2f;

#define TM   64      // rows per block
#define KD   512     // reduction dim
#define HD   512     // hidden size
#define LDA  520     // f16 LDS stride (16B-aligned rows, bank-skewed)
#define LDG  520     // f16 LDS stride for gate buffers

// ---------------------------------------------------------------------------
// Prologue: fp32 weights -> f16 in workspace (weights then live in L2).
// ---------------------------------------------------------------------------
__global__ __launch_bounds__(256) void lngru_cvt_weights(
    const float* __restrict__ Wi, const float* __restrict__ Wh,
    _Float16* __restrict__ Dx, _Float16* __restrict__ Dh)
{
  int i4 = (blockIdx.x * 256 + threadIdx.x) * 4;
  if (i4 < 1536 * 512) {
    float4 a = *(const float4*)(Wi + i4);
    float4 b = *(const float4*)(Wh + i4);
    Dx[i4 + 0] = (_Float16)a.x; Dx[i4 + 1] = (_Float16)a.y;
    Dx[i4 + 2] = (_Float16)a.z; Dx[i4 + 3] = (_Float16)a.w;
    Dh[i4 + 0] = (_Float16)b.x; Dh[i4 + 1] = (_Float16)b.y;
    Dh[i4 + 2] = (_Float16)b.z; Dh[i4 + 3] = (_Float16)b.w;
  }
}

// ---------------------------------------------------------------------------
// Fused LayerNorm-GRU cell, 64 rows per block.
// Chunk order 0 (r gate) -> 2 (n candidate) -> 1 (z gate + output).
// LN stats are reduced from WMMA accumulators (shuffle butterfly + LDS),
// normalized values merge in place into one f16 gate buffer.
// ---------------------------------------------------------------------------
__global__ __launch_bounds__(256) void lngru_fused(
    const float* __restrict__ x,  const float* __restrict__ hx,
    const float* __restrict__ b_i2h, const float* __restrict__ b_h2h,
    const _Float16* __restrict__ Wx, const _Float16* __restrict__ Wh,
    float* __restrict__ out)
{
  __shared__ _Float16 sA[2][TM][LDA];   // staged f16 activations: [0]=x, [1]=hx
  __shared__ _Float16 sG[TM][LDG];      // gate accumulation buffer (nx, then += [r*]nh)
  __shared__ _Float16 sRN[TM][LDG];     // r gate, later overwritten by n candidate
  __shared__ v2f      sPartW[TM][8];    // per-(row, wave) partial {sum, sumsq}
  __shared__ v2f      sStat[TM];        // per-row {mean, rstd}

  const int tid  = threadIdx.x;
  const int wave = tid >> 5;
  const int lane = tid & 31;
  const int rlo  = lane & 15;
  const int hi   = lane >> 4;           // half-wave select
  const int row0 = blockIdx.x * TM;
  const int ncol0 = wave * 64;          // each wave owns 64 of the 512 chunk columns

  // ---- stage x and hx rows to LDS as f16 -------------------------------
  for (int idx = tid; idx < TM * KD / 4; idx += 256) {
    int r  = idx >> 7;
    int c4 = (idx & 127) << 2;
    float4 vx = *(const float4*)(x  + (size_t)(row0 + r) * KD + c4);
    float4 vh = *(const float4*)(hx + (size_t)(row0 + r) * KD + c4);
    sA[0][r][c4 + 0] = (_Float16)vx.x; sA[0][r][c4 + 1] = (_Float16)vx.y;
    sA[0][r][c4 + 2] = (_Float16)vx.z; sA[0][r][c4 + 3] = (_Float16)vx.w;
    sA[1][r][c4 + 0] = (_Float16)vh.x; sA[1][r][c4 + 1] = (_Float16)vh.y;
    sA[1][r][c4 + 2] = (_Float16)vh.z; sA[1][r][c4 + 3] = (_Float16)vh.w;
  }
  __syncthreads();

  const int order[3] = {0, 2, 1};
  for (int ci = 0; ci < 3; ++ci) {
    const int c = order[ci];

    for (int p = 0; p < 2; ++p) {       // p==0: x-path, p==1: h-path
      const _Float16* __restrict__ W    = (p ? Wh : Wx) + (size_t)c * HD * KD;
      const float*    __restrict__ bias = (p ? b_h2h : b_i2h) + c * HD;

      v8f acc[4][4];
      #pragma unroll
      for (int mi = 0; mi < 4; ++mi)
        #pragma unroll
        for (int ni = 0; ni < 4; ++ni) { v8f z = {}; acc[mi][ni] = z; }

      for (int k0 = 0; k0 < KD; k0 += 32) {
        // A fragments (16x32 f16): lane<16 -> K{0..7,16..23}, lane>=16 -> K{8..15,24..31}
        v16h a[4];
        #pragma unroll
        for (int mi = 0; mi < 4; ++mi) {
          const _Float16* ap = &sA[p][mi * 16 + rlo][k0 + hi * 8];
          v8h lo = *(const v8h*)ap;
          v8h up = *(const v8h*)(ap + 16);
          #pragma unroll
          for (int e = 0; e < 8; ++e) { a[mi][e] = lo[e]; a[mi][e + 8] = up[e]; }
        }
        // B fragments (32x16 f16): lane n -> W[n, k0+{0..7,16..23}], lane n+16 -> W[n, k0+{8..15,24..31}]
        #pragma unroll
        for (int ni = 0; ni < 4; ++ni) {
          const _Float16* bp = W + (size_t)(ncol0 + ni * 16 + rlo) * KD + k0 + hi * 8;
          v8h b0 = *(const v8h*)bp;
          v8h b1 = *(const v8h*)(bp + 16);
          v16h b;
          #pragma unroll
          for (int e = 0; e < 8; ++e) { b[e] = b0[e]; b[e + 8] = b1[e]; }
          #pragma unroll
          for (int mi = 0; mi < 4; ++mi)
            acc[mi][ni] = __builtin_amdgcn_wmma_f32_16x16x32_f16(
                false, a[mi], false, b, (short)0, acc[mi][ni], false, false);
        }
      }

      // ---- fold bias into accumulators --------------------------------
      float bv[4];
      #pragma unroll
      for (int ni = 0; ni < 4; ++ni) bv[ni] = bias[ncol0 + ni * 16 + rlo];
      #pragma unroll
      for (int mi = 0; mi < 4; ++mi)
        #pragma unroll
        for (int ni = 0; ni < 4; ++ni)
          #pragma unroll
          for (int i = 0; i < 8; ++i)
            acc[mi][ni][i] += bv[ni];

      // ---- LN partial sums from registers -----------------------------
      // C layout: VGPR i of acc[mi][ni] -> row mi*16+hi*8+i, col ncol0+ni*16+(lane&15).
      #pragma unroll
      for (int mi = 0; mi < 4; ++mi)
        #pragma unroll
        for (int i = 0; i < 8; ++i) {
          float s = 0.f, q = 0.f;
          #pragma unroll
          for (int ni = 0; ni < 4; ++ni) { float v = acc[mi][ni][i]; s += v; q += v * v; }
          #pragma unroll
          for (int m = 1; m < 16; m <<= 1) {      // butterfly across the 16 column-lanes
            s += __shfl_xor(s, m, 32);
            q += __shfl_xor(q, m, 32);
          }
          if (rlo == 0) { v2f pr = {s, q}; sPartW[mi * 16 + hi * 8 + i][wave] = pr; }
        }
      __syncthreads();
      if (tid < TM) {
        float s = 0.f, q = 0.f;
        #pragma unroll
        for (int w = 0; w < 8; ++w) { v2f pr = sPartW[tid][w]; s += pr.x; q += pr.y; }
        float mean = s * (1.0f / 512.0f);
        float var  = q * (1.0f / 512.0f) - mean * mean;
        v2f st = {mean, rsqrtf(var + 1e-5f)};
        sStat[tid] = st;
      }
      __syncthreads();

      // ---- normalize accumulators, merge into gate buffer -------------
      #pragma unroll
      for (int mi = 0; mi < 4; ++mi)
        #pragma unroll
        for (int i = 0; i < 8; ++i) {
          int row = mi * 16 + hi * 8 + i;
          v2f st = sStat[row];
          #pragma unroll
          for (int ni = 0; ni < 4; ++ni) {
            int col = ncol0 + ni * 16 + rlo;
            float nv = (acc[mi][ni][i] - st.x) * st.y;
            if (p == 0) {
              sG[row][col] = (_Float16)nv;                    // nx
            } else if (c == 2) {                              // += r * nh
              sG[row][col] = (_Float16)((float)sG[row][col] + (float)sRN[row][col] * nv);
            } else {                                          // += nh
              sG[row][col] = (_Float16)((float)sG[row][col] + nv);
            }
          }
        }
      __syncthreads();
    }

    // ---- elementwise epilogue for this chunk -------------------------
    for (int idx = tid; idx < TM * HD; idx += 256) {
      int r = idx >> 9, cc = idx & 511;
      float g = (float)sG[r][cc];
      if (c == 0) {
        sRN[r][cc] = (_Float16)(1.0f / (1.0f + __expf(-g)));  // r gate
      } else if (c == 2) {
        sRN[r][cc] = (_Float16)tanhf(g);                      // n candidate (overwrites r)
      } else {
        float zt = 1.0f / (1.0f + __expf(-g));                // z gate
        float nt = (float)sRN[r][cc];
        float hv = hx[(size_t)(row0 + r) * HD + cc];          // f32 reload for precision
        out[(size_t)(row0 + r) * HD + cc] = zt * hv + (1.0f - zt) * nt;
      }
    }
    __syncthreads();
  }
}

// ---------------------------------------------------------------------------
extern "C" void kernel_launch(void* const* d_in, const int* in_sizes, int n_in,
                              void* d_out, int out_size, void* d_ws, size_t ws_size,
                              hipStream_t stream) {
  const float* x  = (const float*)d_in[0];
  const float* hx = (const float*)d_in[1];
  const float* Wi = (const float*)d_in[2];
  const float* bi = (const float*)d_in[3];
  const float* Wh = (const float*)d_in[4];
  const float* bh = (const float*)d_in[5];

  _Float16* Dx = (_Float16*)d_ws;          // 1536*512 f16
  _Float16* Dh = Dx + 1536 * 512;          // 1536*512 f16 (~3 MB of d_ws total)

  lngru_cvt_weights<<<768, 256, 0, stream>>>(Wi, Wh, Dx, Dh);
  lngru_fused<<<16384 / TM, 256, 0, stream>>>(x, hx, bi, bh, Dx, Dh, (float*)d_out);
}
